// SimCLR_28905129902207
// MI455X (gfx1250) — compile-verified, matrix operand order checked by reading, not run on hardware
//
#include <hip/hip_runtime.h>
#include <hip/hip_bf16.h>

typedef __attribute__((ext_vector_type(16))) _Float16 v16h;
typedef __attribute__((ext_vector_type(8)))  _Float16 v8h;
typedef __attribute__((ext_vector_type(8)))  float    v8f;
typedef __attribute__((ext_vector_type(4)))  float    v4f;

constexpr int kN     = 4096;   // per-view batch
constexpr int kFeat  = 2048;
constexpr int kProj  = 128;
constexpr int kTwoN  = 8192;
constexpr float kInvTemp = 2.0f;   // 1 / 0.5
constexpr float kShift   = 2.0f;   // cosine <= 1  =>  sim/temp <= 2
constexpr int kJSplit = 4;

#if defined(__has_builtin)
#if __has_builtin(__builtin_amdgcn_tensor_load_to_lds) && \
    __has_builtin(__builtin_amdgcn_s_wait_tensorcnt)
#define USE_TDM 1
#endif
#endif

// ---------------- conversion kernels ----------------

__global__ void cvt_f32_to_f16(const float* __restrict__ in,
                               _Float16* __restrict__ out, int n) {
  int i = blockIdx.x * blockDim.x + threadIdx.x;
  if (i < n) out[i] = (_Float16)in[i];
}

// in: [R][C] f32 row-major  ->  out: [C][R] f16 (i.e. B^T with contiguous K)
__global__ void transpose_cvt(const float* __restrict__ in,
                              _Float16* __restrict__ out, int R, int C) {
  int i = blockIdx.x * blockDim.x + threadIdx.x;
  if (i < R * C) {
    int r = i / C, c = i % C;
    out[(size_t)c * R + r] = (_Float16)in[i];
  }
}

// ---------------- TDM tile loader ----------------
// Loads a 128x32 f16 tile (rows x K) from a [rows_total][K] f16 tensor into
// LDS with a 40-half (80B) padded pitch: pad_interval = 16 DWORDs of data,
// pad_amount = 4 DWORDs  ->  16 + 4 = 20 DW = 40 halves per row.

constexpr int kLdsPitch = 40;  // padded row pitch in halves, conflict-free

#ifdef USE_TDM
typedef __attribute__((ext_vector_type(4))) unsigned int u32x4;
typedef __attribute__((ext_vector_type(8))) int          i32x8;
typedef __attribute__((ext_vector_type(4))) int          i32x4;

__device__ __forceinline__ void tdm_load_tile_f16(
    unsigned long long gaddr,  // byte address of tile start
    unsigned lds_off,          // LDS byte offset of destination
    unsigned K, unsigned rows) {
  u32x4 g0;
  g0[0] = 1u;  // count=1 valid user descriptor
  g0[1] = lds_off;
  g0[2] = (unsigned)(gaddr & 0xFFFFFFFFull);
  g0[3] = (unsigned)((gaddr >> 32) & 0x01FFFFFFull) | (2u << 30);  // type=2
  i32x8 g1;
  // data_size=1 (2B), pad_enable, pad_interval=3 (16 DW), pad_amount=3 (4 DW)
  g1[0] = (int)((1u << 16) | (1u << 20) | (3u << 22) | (3u << 25));
  g1[1] = (int)((K & 0xFFFFu) << 16);                    // tensor_dim0[15:0]
  g1[2] = (int)((K >> 16) | ((rows & 0xFFFFu) << 16));   // dim0 hi | dim1 lo
  g1[3] = (int)((rows >> 16) | (32u << 16));             // dim1 hi | tile_dim0
  g1[4] = (int)128;                                      // tile_dim1 | tile_dim2=0
  g1[5] = (int)K;                                        // tensor_dim0_stride lo
  g1[6] = 0;
  g1[7] = 0;
  i32x4 z4 = {0, 0, 0, 0};
#if __clang_major__ >= 23
  i32x8 z8 = {0, 0, 0, 0, 0, 0, 0, 0};
  __builtin_amdgcn_tensor_load_to_lds(g0, g1, z4, z4, z8, 0);
#else
  __builtin_amdgcn_tensor_load_to_lds(g0, g1, z4, z4, 0);
#endif
}
#endif  // USE_TDM

// ---------------- WMMA GEMM:  C = act(A @ B + bias) ----------------
// A: [M][K] f16 row-major, BT: [N][K] f16 row-major (i.e. B column-major)
// Block tile 128x128, K-step 32, 256 threads = 8 waves; wave -> 32x64 slab.
// TDM path: double-buffered LDS, wave 0 drives the Tensor Data Mover.

template <bool RELU, bool HALF_OUT>
__global__ __launch_bounds__(256) void gemm_wmma(
    const _Float16* __restrict__ A, const _Float16* __restrict__ BT,
    const float* __restrict__ bias, void* __restrict__ Cout,
    int M, int Nn, int K) {
  __shared__ _Float16 sA[2][128 * kLdsPitch];
  __shared__ _Float16 sB[2][128 * kLdsPitch];

  const int tid  = threadIdx.x;
  const int lane = tid & 31;
  const int wave = tid >> 5;
  const int wr   = wave >> 1;   // 0..3 : 32-row slab
  const int wc   = wave & 1;    // 0..1 : 64-col slab
  const int row0 = blockIdx.x * 128;
  const int col0 = blockIdx.y * 128;

  const int r   = lane & 15;
  const int sel = lane >> 4;

  v8f acc[2][4];
#pragma unroll
  for (int a = 0; a < 2; ++a)
#pragma unroll
    for (int b = 0; b < 4; ++b)
#pragma unroll
      for (int e = 0; e < 8; ++e) acc[a][b][e] = 0.0f;

#ifdef USE_TDM
  const unsigned long long aBase = (unsigned long long)A;
  const unsigned long long bBase = (unsigned long long)BT;
  const unsigned ldsA[2] = {(unsigned)(unsigned long long)&sA[0][0],
                            (unsigned)(unsigned long long)&sA[1][0]};
  const unsigned ldsB[2] = {(unsigned)(unsigned long long)&sB[0][0],
                            (unsigned)(unsigned long long)&sB[1][0]};
  const int nsteps = K / 32;

  if (wave == 0) {  // prologue: stage step 0 into buffer 0
    tdm_load_tile_f16(aBase + (size_t)row0 * K * 2, ldsA[0], (unsigned)K,
                      (unsigned)M);
    tdm_load_tile_f16(bBase + (size_t)col0 * K * 2, ldsB[0], (unsigned)K,
                      (unsigned)Nn);
  }

  for (int step = 0; step < nsteps; ++step) {
    const int cur = step & 1;
    if (wave == 0) __builtin_amdgcn_s_wait_tensorcnt((short)0);
    __syncthreads();  // buffer `cur` ready; buffer `1-cur` reads retired
    if (wave == 0 && step + 1 < nsteps) {
      const size_t kn = (size_t)(step + 1) * 32;
      tdm_load_tile_f16(aBase + ((size_t)row0 * K + kn) * 2, ldsA[1 - cur],
                        (unsigned)K, (unsigned)M);
      tdm_load_tile_f16(bBase + ((size_t)col0 * K + kn) * 2, ldsB[1 - cur],
                        (unsigned)K, (unsigned)Nn);
    }

    v16h af[2], bf[4];
#pragma unroll
    for (int mi = 0; mi < 2; ++mi) {
      const _Float16* p = &sA[cur][(wr * 32 + mi * 16 + r) * kLdsPitch + sel * 8];
      v8h lo = *(const v8h*)p;
      v8h hi = *(const v8h*)(p + 16);
#pragma unroll
      for (int e = 0; e < 8; ++e) { af[mi][e] = lo[e]; af[mi][8 + e] = hi[e]; }
    }
#pragma unroll
    for (int ni = 0; ni < 4; ++ni) {
      const _Float16* p = &sB[cur][(wc * 64 + ni * 16 + r) * kLdsPitch + sel * 8];
      v8h lo = *(const v8h*)p;
      v8h hi = *(const v8h*)(p + 16);
#pragma unroll
      for (int e = 0; e < 8; ++e) { bf[ni][e] = lo[e]; bf[ni][8 + e] = hi[e]; }
    }
#pragma unroll
    for (int mi = 0; mi < 2; ++mi)
#pragma unroll
      for (int ni = 0; ni < 4; ++ni)
        acc[mi][ni] = __builtin_amdgcn_wmma_f32_16x16x32_f16(
            false, af[mi], false, bf[ni], (short)0, acc[mi][ni], false, false);
  }
#else  // manual staging fallback
  const int srow = tid >> 1;         // 0..127
  const int skof = (tid & 1) * 16;   // 0 | 16
  for (int k0 = 0; k0 < K; k0 += 32) {
    const _Float16* ga = A  + (size_t)(row0 + srow) * K + k0 + skof;
    const _Float16* gb = BT + (size_t)(col0 + srow) * K + k0 + skof;
    v8h a0 = *(const v8h*)ga;
    v8h a1 = *(const v8h*)(ga + 8);
    v8h b0 = *(const v8h*)gb;
    v8h b1 = *(const v8h*)(gb + 8);

    __syncthreads();
    *(v8h*)(&sA[0][srow * kLdsPitch + skof])     = a0;
    *(v8h*)(&sA[0][srow * kLdsPitch + skof + 8]) = a1;
    *(v8h*)(&sB[0][srow * kLdsPitch + skof])     = b0;
    *(v8h*)(&sB[0][srow * kLdsPitch + skof + 8]) = b1;
    __syncthreads();

    if (k0 + 32 < K) {
      __builtin_prefetch(ga + 32, 0, 1);
      __builtin_prefetch(gb + 32, 0, 1);
    }

    v16h af[2], bf[4];
#pragma unroll
    for (int mi = 0; mi < 2; ++mi) {
      const _Float16* p = &sA[0][(wr * 32 + mi * 16 + r) * kLdsPitch + sel * 8];
      v8h lo = *(const v8h*)p;
      v8h hi = *(const v8h*)(p + 16);
#pragma unroll
      for (int e = 0; e < 8; ++e) { af[mi][e] = lo[e]; af[mi][8 + e] = hi[e]; }
    }
#pragma unroll
    for (int ni = 0; ni < 4; ++ni) {
      const _Float16* p = &sB[0][(wc * 64 + ni * 16 + r) * kLdsPitch + sel * 8];
      v8h lo = *(const v8h*)p;
      v8h hi = *(const v8h*)(p + 16);
#pragma unroll
      for (int e = 0; e < 8; ++e) { bf[ni][e] = lo[e]; bf[ni][8 + e] = hi[e]; }
    }
#pragma unroll
    for (int mi = 0; mi < 2; ++mi)
#pragma unroll
      for (int ni = 0; ni < 4; ++ni)
        acc[mi][ni] = __builtin_amdgcn_wmma_f32_16x16x32_f16(
            false, af[mi], false, bf[ni], (short)0, acc[mi][ni], false, false);
  }
#endif

  // epilogue: lane holds col n = r, rows v + 8*sel of each 16x16 tile
#pragma unroll
  for (int mi = 0; mi < 2; ++mi)
#pragma unroll
    for (int ni = 0; ni < 4; ++ni) {
      const int gcol = col0 + wc * 64 + ni * 16 + r;
      const float bv = bias[gcol];
#pragma unroll
      for (int v = 0; v < 8; ++v) {
        const int grow = row0 + wr * 32 + mi * 16 + v + 8 * sel;
        float xv = acc[mi][ni][v] + bv;
        if (RELU) xv = xv > 0.0f ? xv : 0.0f;
        if (HALF_OUT)
          ((_Float16*)Cout)[(size_t)grow * Nn + gcol] = (_Float16)xv;
        else
          ((float*)Cout)[(size_t)grow * Nn + gcol] = xv;
      }
    }
}

// ---------------- row L2-normalize: Z f32 [2N][128] -> Zn f16 ----------------

__global__ void rownorm(const float* __restrict__ Z, _Float16* __restrict__ Zn) {
  const int row  = blockIdx.x;
  const int lane = threadIdx.x;  // 32 threads, 4 elems each
  const float* zp = Z + (size_t)row * kProj + lane * 4;
  v4f v = *(const v4f*)zp;
  float ss = v.x * v.x + v.y * v.y + v.z * v.z + v.w * v.w;
#pragma unroll
  for (int d = 16; d >= 1; d >>= 1) ss += __shfl_xor(ss, d, 32);
  const float inv = rsqrtf(ss);
  _Float16* op = Zn + (size_t)row * kProj + lane * 4;
  op[0] = (_Float16)(v.x * inv);
  op[1] = (_Float16)(v.y * inv);
  op[2] = (_Float16)(v.z * inv);
  op[3] = (_Float16)(v.w * inv);
}

// ---------------- zero-init accumulators ----------------

__global__ void zero_accum(float* __restrict__ out, float* __restrict__ swsum,
                           float* __restrict__ spos) {
  const int i = blockIdx.x * blockDim.x + threadIdx.x;
  if (i == 0) *out = 0.0f;
  if (i < kTwoN) { swsum[i] = 0.0f; spos[i] = 0.0f; }
}

// ---------------- similarity + masked sum-exp partials ----------------
// sim = Zn @ Zn^T ; BT operand is Zn itself. One wave per 16-row i-tile,
// j split kJSplit ways across blockIdx.y. Fixed shift (s <= 2) replaces
// online max: sumexp += exp(s - 2), self masked, positive recorded.

__global__ __launch_bounds__(256) void sim_partial(
    const _Float16* __restrict__ Zn, float* __restrict__ swsum,
    float* __restrict__ spos) {
  const int lane  = threadIdx.x & 31;
  const int wave  = threadIdx.x >> 5;
  const int itile = blockIdx.x * 8 + wave;  // 0..511
  const int i0    = itile * 16;
  const int r     = lane & 15;
  const int sel   = lane >> 4;

  v16h af[4];
#pragma unroll
  for (int kc = 0; kc < 4; ++kc) {
    const _Float16* p = Zn + (size_t)(i0 + r) * kProj + kc * 32 + sel * 8;
    v8h lo = *(const v8h*)p;
    v8h hi = *(const v8h*)(p + 16);
#pragma unroll
    for (int e = 0; e < 8; ++e) { af[kc][e] = lo[e]; af[kc][8 + e] = hi[e]; }
  }

  float se[8], ps[8];
#pragma unroll
  for (int v = 0; v < 8; ++v) { se[v] = 0.0f; ps[v] = 0.0f; }

  const int jt0 = blockIdx.y * (512 / kJSplit);
  for (int jt = jt0; jt < jt0 + 512 / kJSplit; ++jt) {
    const int j0 = jt * 16;
    v8f acc;
#pragma unroll
    for (int e = 0; e < 8; ++e) acc[e] = 0.0f;
#pragma unroll
    for (int kc = 0; kc < 4; ++kc) {
      const _Float16* p = Zn + (size_t)(j0 + r) * kProj + kc * 32 + sel * 8;
      v8h lo = *(const v8h*)p;
      v8h hi = *(const v8h*)(p + 16);
      v16h bf;
#pragma unroll
      for (int e = 0; e < 8; ++e) { bf[e] = lo[e]; bf[8 + e] = hi[e]; }
      acc = __builtin_amdgcn_wmma_f32_16x16x32_f16(
          false, af[kc], false, bf, (short)0, acc, false, false);
    }
    const int j = j0 + r;
#pragma unroll
    for (int v = 0; v < 8; ++v) {
      const int i = i0 + v + 8 * sel;
      const float s = acc[v] * kInvTemp;
      float e = __expf(s - kShift);
      if (j == i) e = 0.0f;                           // mask self
      if (j == ((i + kN) & (kTwoN - 1))) ps[v] += s;  // positive logit
      se[v] += e;
    }
  }

  // reduce across the 16 lanes of each half-wave (xor over bits 0..3)
#pragma unroll
  for (int v = 0; v < 8; ++v) {
    float a = se[v], b = ps[v];
#pragma unroll
    for (int d = 1; d < 16; d <<= 1) {
      a += __shfl_xor(a, d, 32);
      b += __shfl_xor(b, d, 32);
    }
    if (r == v) {
      atomicAdd(&swsum[i0 + v + 8 * sel], a);
      atomicAdd(&spos[i0 + v + 8 * sel], b);
    }
  }
}

// ---------------- finalize: loss = mean(log(sumexp)+shift - pos) ----------------

__global__ void finalize(const float* __restrict__ swsum,
                         const float* __restrict__ spos,
                         float* __restrict__ out) {
  const int i = blockIdx.x * 256 + threadIdx.x;
  float c = 0.0f;
  if (i < kTwoN)
    c = (__logf(swsum[i]) + kShift - spos[i]) * (1.0f / (float)kTwoN);
  __shared__ float red[8];
#pragma unroll
  for (int d = 16; d >= 1; d >>= 1) c += __shfl_xor(c, d, 32);
  const int lane = threadIdx.x & 31, w = threadIdx.x >> 5;
  if (lane == 0) red[w] = c;
  __syncthreads();
  if (w == 0) {
    float t = lane < 8 ? red[lane] : 0.0f;
#pragma unroll
    for (int d = 4; d >= 1; d >>= 1) t += __shfl_xor(t, d, 32);
    if (lane == 0) atomicAdd(out, t);
  }
}

// ---------------- host launcher ----------------

extern "C" void kernel_launch(void* const* d_in, const int* in_sizes, int n_in,
                              void* d_out, int out_size, void* d_ws,
                              size_t ws_size, hipStream_t stream) {
  const float* x  = (const float*)d_in[0];
  const float* y  = (const float*)d_in[1];
  const float* W1 = (const float*)d_in[2];
  const float* b1 = (const float*)d_in[3];
  const float* W2 = (const float*)d_in[4];
  const float* b2 = (const float*)d_in[5];
  float* out = (float*)d_out;

  char* ws = (char*)d_ws;
  size_t off = 0;
  auto alloc = [&](size_t bytes) {
    void* p = ws + off;
    off = (off + bytes + 255) & ~(size_t)255;
    return p;
  };
  _Float16* Ah   = (_Float16*)alloc((size_t)kTwoN * kFeat * 2);  // [2N][F] f16
  _Float16* W1T  = (_Float16*)alloc((size_t)kFeat * kFeat * 2);  // [F][F]
  _Float16* W2T  = (_Float16*)alloc((size_t)kProj * kFeat * 2);  // [P][F]
  _Float16* Hh   = (_Float16*)alloc((size_t)kTwoN * kFeat * 2);  // [2N][F]
  float*    Zf   = (float*)   alloc((size_t)kTwoN * kProj * 4);  // [2N][P]
  _Float16* Znh  = (_Float16*)alloc((size_t)kTwoN * kProj * 2);  // [2N][P]
  float*    swsum= (float*)   alloc((size_t)kTwoN * 4);
  float*    spos = (float*)   alloc((size_t)kTwoN * 4);
  (void)off; (void)ws_size; (void)in_sizes; (void)n_in; (void)out_size;

  const int nView = kN * kFeat;
  cvt_f32_to_f16<<<nView / 256, 256, 0, stream>>>(x, Ah, nView);
  cvt_f32_to_f16<<<nView / 256, 256, 0, stream>>>(y, Ah + (size_t)nView, nView);
  transpose_cvt<<<(kFeat * kFeat) / 256, 256, 0, stream>>>(W1, W1T, kFeat, kFeat);
  transpose_cvt<<<(kFeat * kProj) / 256, 256, 0, stream>>>(W2, W2T, kFeat, kProj);

  // H = relu(A @ W1 + b1)  -> f16
  gemm_wmma<true, true><<<dim3(kTwoN / 128, kFeat / 128), 256, 0, stream>>>(
      Ah, W1T, b1, (void*)Hh, kTwoN, kFeat, kFeat);
  // Z = H @ W2 + b2        -> f32
  gemm_wmma<false, false><<<dim3(kTwoN / 128, kProj / 128), 256, 0, stream>>>(
      Hh, W2T, b2, (void*)Zf, kTwoN, kProj, kFeat);

  rownorm<<<kTwoN, 32, 0, stream>>>(Zf, Znh);
  zero_accum<<<kTwoN / 256, 256, 0, stream>>>(out, swsum, spos);
  sim_partial<<<dim3(kTwoN / 16 / 8, kJSplit), 256, 0, stream>>>(Znh, swsum, spos);
  finalize<<<kTwoN / 256, 256, 0, stream>>>(swsum, spos, out);
}
